// AKTInt_64613488001463
// MI455X (gfx1250) — matrix operand order repaired and checked by reading, not compiled, not run on hardware
//
#include <hip/hip_runtime.h>

// ---------------------------------------------------------------- types
typedef __bf16 bf16_t;
typedef __attribute__((ext_vector_type(16))) __bf16 v16bf;
typedef __attribute__((ext_vector_type(8)))  float  v8f;
typedef unsigned int u32x4 __attribute__((ext_vector_type(4)));
typedef int i32x4_t __attribute__((ext_vector_type(4)));
typedef int i32x8_t __attribute__((ext_vector_type(8)));

#define DEV __device__ __forceinline__

static constexpr int BSZ  = 16;
static constexpr int SEQ  = 512;
static constexpr int DMOD = 512;
static constexpr int HDS  = 8;
static constexpr int DK   = 64;
static constexpr int MTOT = BSZ * SEQ;          // 8192 rows

// attention LDS layout (bytes)
static constexpr int VT_LD      = 520;          // padded Vt row (bf16 elems)
static constexpr size_t OFF_VT  = 73728;        // after s_q(8KB)+s_k(64KB)
static constexpr size_t OFF_SC  = OFF_VT + (size_t)64 * VT_LD * 2; // 140288
static constexpr size_t ATTN_LDS = OFF_SC + (size_t)64 * 512 * 4;  // 271360

// ---------------------------------------------------------------- helpers
DEV bf16_t to_bf16(float f) {
  union { float f; unsigned u; } a; a.f = f;
  unsigned r = a.u + 0x7fffu + ((a.u >> 16) & 1u);   // round-nearest-even
  unsigned short h = (unsigned short)(r >> 16);
  return __builtin_bit_cast(bf16_t, h);
}

// A fragment, 16x32 bf16: lane = (kg<<4)|m ; elems 0..7 -> K=kg*8+e, 8..15 -> K=16+kg*8+(e-8)
DEV v16bf load_frag_a(const bf16_t* rowk /* row base + kbase */, int kg) {
  union { v16bf v; u32x4 q[2]; } f;
  f.q[0] = *(const u32x4*)(rowk + kg * 8);
  f.q[1] = *(const u32x4*)(rowk + kg * 8 + 16);
  return f.v;
}
// B fragment, 32x16 bf16: lane = (kg<<4)|n ; elems 0..7 -> K=kg*8+e, 8..15 -> K=16+kg*8+(e-8)
DEV v16bf load_frag_b(const bf16_t* rowk /* col-row base + kbase */, int kg) {
  union { v16bf v; u32x4 q[2]; } f;
  f.q[0] = *(const u32x4*)(rowk + kg * 8);
  f.q[1] = *(const u32x4*)(rowk + kg * 8 + 16);
  return f.v;
}
DEV v8f wmma_bf16(v16bf a, v16bf b, v8f c) {
  return __builtin_amdgcn_wmma_f32_16x16x32_bf16(false, a, false, b, (short)0, c,
                                                 false, false);
}
DEV float wave_red_max(float v) {
#pragma unroll
  for (int m = 16; m >= 1; m >>= 1) v = fmaxf(v, __shfl_xor(v, m, 32));
  return v;
}
DEV float wave_red_sum(float v) {
#pragma unroll
  for (int m = 16; m >= 1; m >>= 1) v += __shfl_xor(v, m, 32);
  return v;
}
DEV float wave_incl_scan(float v, int lane) {
#pragma unroll
  for (int d = 1; d < 32; d <<= 1) {
    float t = __shfl_up(v, d, 32);
    if (lane >= d) v += t;
  }
  return v;
}

// --------------------------------------------------------- TDM global->LDS
// 1D contiguous copy of nelem bf16 elements, expressed as a 2D (nelem x 1)
// tile per CDNA5 ISA D# groups 0/1. Issued once per wave.
#if __has_builtin(__builtin_amdgcn_tensor_load_to_lds)
#define HAVE_TDM 1
DEV void tdm_load_1d_bf16(unsigned lds_off, const bf16_t* gptr, unsigned nelem) {
  unsigned long long ga = (unsigned long long)gptr;
  u32x4 g0;
  g0[0] = 1u;                                    // count=1, user descriptor
  g0[1] = lds_off;                               // lds_addr (bytes)
  g0[2] = (unsigned)ga;                          // global_addr[31:0]
  g0[3] = ((unsigned)(ga >> 32) & 0x01FFFFFFu)   // global_addr[56:32]
          | (2u << 30);                          // type=2 ("image")
  i32x8_t g1;
  g1[0] = (int)0x00010000u;                      // workgroup_mask=0, data_size=1(2B)
  g1[1] = (int)((nelem & 0xFFFFu) << 16);        // tensor_dim0[15:0]
  g1[2] = (int)(((nelem >> 16) & 0xFFFFu) | (1u << 16)); // dim0 hi | tensor_dim1=1
  g1[3] = (int)((nelem & 0xFFFFu) << 16);        // tensor_dim1 hi=0 | tile_dim0
  g1[4] = 1;                                     // tile_dim1=1, tile_dim2=0
  g1[5] = (int)nelem;                            // tensor_dim0_stride[31:0]
  g1[6] = 0;                                     // stride hi | dim1_stride lo
  g1[7] = 0;
  i32x4_t gz = (i32x4_t)0;
#if defined(__clang_major__) && (__clang_major__ >= 23)
  i32x8_t z8 = (i32x8_t)0;
  __builtin_amdgcn_tensor_load_to_lds(g0, g1, gz, gz, z8, 0);
#else
  __builtin_amdgcn_tensor_load_to_lds(g0, g1, gz, gz, 0);
#endif
}
DEV void tdm_wait0() {
#if __has_builtin(__builtin_amdgcn_s_wait_tensorcnt)
  __builtin_amdgcn_s_wait_tensorcnt(0);
#else
  asm volatile("s_wait_tensorcnt 0x0" ::: "memory");
#endif
}
#else
#define HAVE_TDM 0
#endif

// ---------------------------------------------------------------- kernels
__global__ void stats_kernel(const int* __restrict__ sgap, const int* __restrict__ pcount,
                             int n, float* __restrict__ stats) {
  __shared__ float sh[4][8];
  const int tid = threadIdx.x, w = tid >> 5, lane = tid & 31;
  float a0 = 0.f, a1 = 0.f, b0 = 0.f, b1 = 0.f;
  for (int i = tid; i < n; i += 256) {
    float x = (float)sgap[i];   a0 += x; a1 += x * x;
    float y = (float)pcount[i]; b0 += y; b1 += y * y;
  }
  a0 = wave_red_sum(a0); a1 = wave_red_sum(a1);
  b0 = wave_red_sum(b0); b1 = wave_red_sum(b1);
  if (lane == 0) { sh[0][w] = a0; sh[1][w] = a1; sh[2][w] = b0; sh[3][w] = b1; }
  __syncthreads();
  if (tid == 0) {
    float A0 = 0, A1 = 0, B0 = 0, B1 = 0;
    for (int i = 0; i < 8; ++i) { A0 += sh[0][i]; A1 += sh[1][i]; B0 += sh[2][i]; B1 += sh[3][i]; }
    float nf = (float)n;
    float ms = A0 / nf, mp = B0 / nf;
    float vs = (A1 - nf * ms * ms) / (nf - 1.f);
    float vp = (B1 - nf * mp * mp) / (nf - 1.f);
    float ss = sqrtf(fmaxf(vs, 0.f)) + 1e-6f;
    float sp = sqrtf(fmaxf(vp, 0.f)) + 1e-6f;
    stats[0] = ms; stats[1] = 1.f / ss; stats[2] = mp; stats[3] = 1.f / sp;
  }
}

__global__ void cvt_kernel(const float* __restrict__ src, bf16_t* __restrict__ dst, int n) {
  int i = blockIdx.x * 256 + threadIdx.x;
  if (i < n) dst[i] = to_bf16(src[i]);
}

// C[M,N] = A[M,K] @ W[N,K]^T ; wave -> 16x64 tile, K unrolled by 64 so all
// 10 fragments are live before the 8 WMMAs (one loadcnt wait per group).
// mode 0: out bf16 head-split [b,h,s,dk], val=(acc+bias)*scale
// mode 1: out fp32 row-major [M,N],      val= acc+bias
__global__ __launch_bounds__(256) void gemm_bf16_kernel(
    const bf16_t* __restrict__ A, const bf16_t* __restrict__ W,
    const float* __restrict__ bias, float scale, int M, int N, int K, int mode,
    float* __restrict__ outF, bf16_t* __restrict__ outH, int S, int dk) {
  const int tid = threadIdx.x;
  const int w = tid >> 5, lane = tid & 31;
  const int kg = lane >> 4, ln = lane & 15;
  const int m0 = blockIdx.x * 128 + w * 16;
  const int n0 = blockIdx.y * 64;

  v8f acc[4] = {v8f{}, v8f{}, v8f{}, v8f{}};
  const bf16_t* arow = A + (size_t)(m0 + ln) * K;
  const bf16_t* wr0 = W + (size_t)(n0 + ln) * K;
  const bf16_t* wr1 = wr0 + (size_t)16 * K;
  const bf16_t* wr2 = wr0 + (size_t)32 * K;
  const bf16_t* wr3 = wr0 + (size_t)48 * K;

  for (int kb = 0; kb < K; kb += 64) {
    v16bf af0 = load_frag_a(arow + kb, kg);
    v16bf af1 = load_frag_a(arow + kb + 32, kg);
    v16bf b00 = load_frag_b(wr0 + kb, kg);
    v16bf b01 = load_frag_b(wr1 + kb, kg);
    v16bf b02 = load_frag_b(wr2 + kb, kg);
    v16bf b03 = load_frag_b(wr3 + kb, kg);
    v16bf b10 = load_frag_b(wr0 + kb + 32, kg);
    v16bf b11 = load_frag_b(wr1 + kb + 32, kg);
    v16bf b12 = load_frag_b(wr2 + kb + 32, kg);
    v16bf b13 = load_frag_b(wr3 + kb + 32, kg);
    acc[0] = wmma_bf16(af0, b00, acc[0]);
    acc[1] = wmma_bf16(af0, b01, acc[1]);
    acc[2] = wmma_bf16(af0, b02, acc[2]);
    acc[3] = wmma_bf16(af0, b03, acc[3]);
    acc[0] = wmma_bf16(af1, b10, acc[0]);
    acc[1] = wmma_bf16(af1, b11, acc[1]);
    acc[2] = wmma_bf16(af1, b12, acc[2]);
    acc[3] = wmma_bf16(af1, b13, acc[3]);
  }
  const int Hn = N / dk;
#pragma unroll
  for (int t = 0; t < 4; ++t) {
    const int col = n0 + t * 16 + ln;
    const float bv = bias[col];
#pragma unroll
    for (int i = 0; i < 8; ++i) {
      const int row = m0 + kg * 8 + i;
      float v = acc[t][i] + bv;
      if (mode == 0) {
        v *= scale;
        int bb = row / S, s = row - bb * S;
        int hh = col / dk, d = col - hh * dk;
        outH[(((size_t)bb * Hn + hh) * S + s) * dk + d] = to_bf16(v);
      } else {
        outF[(size_t)row * N + col] = v;
      }
    }
  }
}

__global__ __launch_bounds__(256) void attn_kernel(
    const bf16_t* __restrict__ Qh, const bf16_t* __restrict__ Kh,
    const bf16_t* __restrict__ Vh, const float* __restrict__ gammas,
    float* __restrict__ X) {
  extern __shared__ char smem[];
  bf16_t* s_q    = (bf16_t*)smem;                 // [64*64]
  bf16_t* s_k    = (bf16_t*)(smem + 8192);        // [512*64]
  bf16_t* s_attn = (bf16_t*)smem;                 // alias [64*512] (after scores done)
  bf16_t* s_vt   = (bf16_t*)(smem + OFF_VT);      // [64*520]
  float*  s_sc   = (float*)(smem + OFF_SC);       // [64*512]

  const int tid = threadIdx.x;
  const int w = tid >> 5, lane = tid & 31;
  const int kg = lane >> 4, ln = lane & 15;
  const int chunk = blockIdx.x & 7;
  const int bh = blockIdx.x >> 3;
  const int b = bh >> 3, h = bh & 7;
  const int row_base = chunk * 64;

  // ---- stage Q chunk + K via TDM (wave 0), V^T transpose by all waves
  const bf16_t* Qp = Qh + ((size_t)bh * SEQ + row_base) * DK;
  const bf16_t* Kp = Kh + (size_t)bh * SEQ * DK;
  const bf16_t* Vp = Vh + (size_t)bh * SEQ * DK;
#if HAVE_TDM
  if (w == 0) {
    tdm_load_1d_bf16((unsigned)(unsigned long long)s_q, Qp, 64 * DK);
    tdm_load_1d_bf16((unsigned)(unsigned long long)s_k, Kp, SEQ * DK);
  }
#else
  {
    const u32x4* qs = (const u32x4*)Qp;
    u32x4* qd = (u32x4*)s_q;
    for (int i = tid; i < (64 * DK) / 8; i += 256) qd[i] = qs[i];
    const u32x4* ks = (const u32x4*)Kp;
    u32x4* kd = (u32x4*)s_k;
    for (int i = tid; i < (SEQ * DK) / 8; i += 256) kd[i] = ks[i];
  }
#endif
  for (int idx = tid; idx < SEQ * DK; idx += 256) {
    int k = idx >> 6, n = idx & 63;
    s_vt[n * VT_LD + k] = Vp[idx];
  }
#if HAVE_TDM
  if (w == 0) tdm_wait0();
#endif
  __syncthreads();

  // ---- scores: Q[64,64] @ K^T -> s_sc[64,512] (pre-scaled via Q)
  {
    const int mt = w >> 1, half = w & 1;
    const int m0 = mt * 16;
    const bf16_t* arow = s_q + (m0 + ln) * DK;
    for (int nt = 0; nt < 16; ++nt) {
      const int n0 = half * 256 + nt * 16;
      const bf16_t* brow = s_k + (n0 + ln) * DK;
      v16bf af0 = load_frag_a(arow, kg);
      v16bf af1 = load_frag_a(arow + 32, kg);
      v16bf bf0 = load_frag_b(brow, kg);
      v16bf bf1 = load_frag_b(brow + 32, kg);
      v8f acc = {};
      acc = wmma_bf16(af0, bf0, acc);
      acc = wmma_bf16(af1, bf1, acc);
#pragma unroll
      for (int i = 0; i < 8; ++i)
        s_sc[(m0 + kg * 8 + i) * SEQ + n0 + ln] = acc[i];
    }
  }
  __syncthreads();

  // ---- per-row: masked softmax -> cumsum -> temporal decay -> softmax -> bf16 attn
  const float g = gammas[h];
  const float gam = -(fmaxf(g, 0.f) + log1pf(expf(-fabsf(g))));  // -softplus

  for (int r = w; r < 64; r += 8) {
    const int iq = row_base + r;             // query index (wave-uniform)
    if (iq == 0) {                           // zero_pad of first query position
      for (int c = 0; c < 16; ++c) s_attn[r * SEQ + c * 32 + lane] = to_bf16(0.f);
      continue;
    }
    float f[16], p[16];
    float m1 = -3.0e38f;
#pragma unroll
    for (int c = 0; c < 16; ++c) {
      int j = c * 32 + lane;
      f[c] = s_sc[r * SEQ + j];
      if (j < iq) m1 = fmaxf(m1, f[c]);
    }
    m1 = wave_red_max(m1);
    float sum = 0.f;
#pragma unroll
    for (int c = 0; c < 16; ++c) {
      int j = c * 32 + lane;
      p[c] = (j < iq) ? expf(f[c] - m1) : 0.f;
      sum += p[c];
    }
    sum = wave_red_sum(sum);
    const float rinv = 1.f / sum;            // disttotal == 1 after normalize
    float run = 0.f;
    const float fi = (float)iq;
    for (int c = 0; c < 16; ++c) {
      int j = c * 32 + lane;
      float incl = wave_incl_scan(p[c] * rinv, lane);
      float cum = run + incl;
      run += __shfl(incl, 31, 32);
      float pos = fabsf(fi - (float)j);
      float dist = sqrtf(fmaxf((1.0f - cum) * pos, 0.f));
      float tempo = fminf(fmaxf(expf(dist * gam), 1e-5f), 1e5f);
      p[c] = (j < iq) ? f[c] * tempo : -3.0e38f;
    }
    float m2 = -3.0e38f;
#pragma unroll
    for (int c = 0; c < 16; ++c) m2 = fmaxf(m2, p[c]);
    m2 = wave_red_max(m2);
    float s2 = 0.f;
#pragma unroll
    for (int c = 0; c < 16; ++c) {
      int j = c * 32 + lane;
      f[c] = (j < iq) ? expf(p[c] - m2) : 0.f;
      s2 += f[c];
    }
    s2 = wave_red_sum(s2);
    const float r2 = 1.f / s2;
#pragma unroll
    for (int c = 0; c < 16; ++c)
      s_attn[r * SEQ + c * 32 + lane] = to_bf16(f[c] * r2);
  }
  __syncthreads();

  // ---- X[64,64] = attn[64,512] @ V[512,64]  (K unrolled by 64)
#pragma unroll
  for (int tt = 0; tt < 2; ++tt) {
    const int tile = w * 2 + tt;
    const int m0 = (tile >> 2) * 16;
    const int n0 = (tile & 3) * 16;
    v8f acc = {};
    const bf16_t* arow = s_attn + (m0 + ln) * SEQ;
    const bf16_t* brow = s_vt + (n0 + ln) * VT_LD;
    for (int kb = 0; kb < SEQ; kb += 64) {
      v16bf af0 = load_frag_a(arow + kb, kg);
      v16bf af1 = load_frag_a(arow + kb + 32, kg);
      v16bf bf0 = load_frag_b(brow + kb, kg);
      v16bf bf1 = load_frag_b(brow + kb + 32, kg);
      acc = wmma_bf16(af0, bf0, acc);
      acc = wmma_bf16(af1, bf1, acc);
    }
#pragma unroll
    for (int i = 0; i < 8; ++i) {
      const int srow = row_base + m0 + kg * 8 + i;
      X[((size_t)b * SEQ + srow) * DMOD + h * DK + n0 + ln] = acc[i];
    }
  }
}

__global__ __launch_bounds__(256) void addnorm_kernel(
    const float* __restrict__ X, const int* __restrict__ sgap,
    const int* __restrict__ pcount, const float* __restrict__ stats,
    const float* __restrict__ Wi1, const float* __restrict__ bi1,
    const float* __restrict__ Wi2, const float* __restrict__ bi2,
    const float* __restrict__ iscale, const float* __restrict__ ln_g,
    const float* __restrict__ ln_b, bf16_t* __restrict__ Yn) {
  __shared__ float s_hid[8][128];
  const int tid = threadIdx.x, w = tid >> 5, lane = tid & 31;
  const int row = blockIdx.x * 8 + w;
  const float ii0 = (tanhf(((float)sgap[row]  - stats[0]) * stats[1]) + 1.f) * 0.5f;
  const float ii1 = (tanhf(((float)pcount[row] - stats[2]) * stats[3]) + 1.f) * 0.5f;
#pragma unroll
  for (int hh = 0; hh < 4; ++hh) {
    int hI = hh * 32 + lane;
    float v = Wi1[hI * 2 + 0] * ii0 + Wi1[hI * 2 + 1] * ii1 + bi1[hI];
    s_hid[w][hI] = fmaxf(v, 0.f);
  }
  __syncthreads();
  const float isc = iscale[0];
  float y[16];
  float sum = 0.f, sq = 0.f;
#pragma unroll
  for (int dd = 0; dd < 16; ++dd) {
    int d = dd * 32 + lane;
    float feat = bi2[d];
    for (int hI = 0; hI < 128; ++hI) feat += Wi2[d * 128 + hI] * s_hid[w][hI];
    float v = X[(size_t)row * DMOD + d] + isc * feat;
    y[dd] = v; sum += v; sq += v * v;
  }
  sum = wave_red_sum(sum); sq = wave_red_sum(sq);
  const float mu = sum * (1.f / DMOD);
  const float var = sq * (1.f / DMOD) - mu * mu;
  const float rs = rsqrtf(var + 1e-5f);
#pragma unroll
  for (int dd = 0; dd < 16; ++dd) {
    int d = dd * 32 + lane;
    float v = (y[dd] - mu) * rs * ln_g[d] + ln_b[d];
    Yn[(size_t)row * DMOD + d] = to_bf16(v);
  }
}

// ---------------------------------------------------------------- launch
extern "C" void kernel_launch(void* const* d_in, const int* in_sizes, int n_in,
                              void* d_out, int out_size, void* d_ws, size_t ws_size,
                              hipStream_t stream) {
  const float* q      = (const float*)d_in[0];
  const float* k      = (const float*)d_in[1];
  const float* v      = (const float*)d_in[2];
  const int*   sgap   = (const int*)d_in[3];
  const int*   pcount = (const int*)d_in[4];
  const float* Wq = (const float*)d_in[5];  const float* bq = (const float*)d_in[6];
  const float* Wk = (const float*)d_in[7];  const float* bk = (const float*)d_in[8];
  const float* Wv = (const float*)d_in[9];  const float* bv = (const float*)d_in[10];
  const float* Wo = (const float*)d_in[11]; const float* bo = (const float*)d_in[12];
  const float* gammas = (const float*)d_in[13];
  const float* ln_g = (const float*)d_in[14];
  const float* ln_b = (const float*)d_in[15];
  const float* Wi1 = (const float*)d_in[16]; const float* bi1 = (const float*)d_in[17];
  const float* Wi2 = (const float*)d_in[18]; const float* bi2 = (const float*)d_in[19];
  const float* iscale = (const float*)d_in[20];

  char* ws = (char*)d_ws;
  size_t off = 0;
  auto alloc = [&](size_t bytes) -> char* {
    char* p = ws + off;
    off += (bytes + 255) & ~(size_t)255;
    return p;
  };
  float*  stats = (float*)alloc(4 * sizeof(float));
  bf16_t* qb  = (bf16_t*)alloc((size_t)MTOT * DMOD * 2);
  bf16_t* kb  = (bf16_t*)alloc((size_t)MTOT * DMOD * 2);
  bf16_t* vb  = (bf16_t*)alloc((size_t)MTOT * DMOD * 2);
  bf16_t* Wqb = (bf16_t*)alloc((size_t)DMOD * DMOD * 2);
  bf16_t* Wkb = (bf16_t*)alloc((size_t)DMOD * DMOD * 2);
  bf16_t* Wvb = (bf16_t*)alloc((size_t)DMOD * DMOD * 2);
  bf16_t* Wob = (bf16_t*)alloc((size_t)DMOD * DMOD * 2);
  bf16_t* Qh  = (bf16_t*)alloc((size_t)MTOT * DMOD * 2);
  bf16_t* Kh  = (bf16_t*)alloc((size_t)MTOT * DMOD * 2);
  bf16_t* Vh  = (bf16_t*)alloc((size_t)MTOT * DMOD * 2);
  float*  X   = (float*)alloc((size_t)MTOT * DMOD * 4);
  bf16_t* Yn  = (bf16_t*)alloc((size_t)MTOT * DMOD * 2);

  (void)hipFuncSetAttribute((const void*)attn_kernel,
                            hipFuncAttributeMaxDynamicSharedMemorySize,
                            (int)ATTN_LDS);

  stats_kernel<<<1, 256, 0, stream>>>(sgap, pcount, MTOT, stats);

  const int nAct = MTOT * DMOD, nW = DMOD * DMOD;
  cvt_kernel<<<(nAct + 255) / 256, 256, 0, stream>>>(q,  qb,  nAct);
  cvt_kernel<<<(nAct + 255) / 256, 256, 0, stream>>>(k,  kb,  nAct);
  cvt_kernel<<<(nAct + 255) / 256, 256, 0, stream>>>(v,  vb,  nAct);
  cvt_kernel<<<(nW + 255) / 256, 256, 0, stream>>>(Wq, Wqb, nW);
  cvt_kernel<<<(nW + 255) / 256, 256, 0, stream>>>(Wk, Wkb, nW);
  cvt_kernel<<<(nW + 255) / 256, 256, 0, stream>>>(Wv, Wvb, nW);
  cvt_kernel<<<(nW + 255) / 256, 256, 0, stream>>>(Wo, Wob, nW);

  dim3 ggrid(MTOT / 128, DMOD / 64);
  const float qscale = 0.125f;   // 1/sqrt(64)
  gemm_bf16_kernel<<<ggrid, 256, 0, stream>>>(qb, Wqb, bq, qscale, MTOT, DMOD, DMOD,
                                              0, nullptr, Qh, SEQ, DK);
  gemm_bf16_kernel<<<ggrid, 256, 0, stream>>>(kb, Wkb, bk, 1.0f, MTOT, DMOD, DMOD,
                                              0, nullptr, Kh, SEQ, DK);
  gemm_bf16_kernel<<<ggrid, 256, 0, stream>>>(vb, Wvb, bv, 1.0f, MTOT, DMOD, DMOD,
                                              0, nullptr, Vh, SEQ, DK);

  attn_kernel<<<BSZ * HDS * (SEQ / 64), 256, ATTN_LDS, stream>>>(Qh, Kh, Vh, gammas, X);

  addnorm_kernel<<<MTOT / 8, 256, 0, stream>>>(X, sgap, pcount, stats, Wi1, bi1,
                                               Wi2, bi2, iscale, ln_g, ln_b, Yn);

  gemm_bf16_kernel<<<ggrid, 256, 0, stream>>>(Yn, Wob, bo, 1.0f, MTOT, DMOD, DMOD,
                                              1, (float*)d_out, nullptr, SEQ, DK);
}
